// IBNModel_57148834840930
// MI455X (gfx1250) — matrix operand order repaired and checked by reading, not compiled, not run on hardware
//
#include <hip/hip_runtime.h>

// Problem constants (match reference)
#define B_    16
#define H_    64
#define N_    512
#define C_    3
#define E_    128
#define D_    32
#define LOUT_ 12
#define KAUG_ 160   // 128 (state) + 2 x 16-row msg slots (double-buffered W|bias regions)

typedef __attribute__((ext_vector_type(16))) __bf16        v16bf;
typedef __attribute__((ext_vector_type(8)))  float         v8f;
typedef __attribute__((ext_vector_type(4)))  unsigned int  v4u;

union FragBF16 { v16bf v; v4u q[2]; };

__device__ __forceinline__ unsigned short f2bf(float x) {
  return (unsigned short)(__float_as_uint(x) >> 16);
}
__device__ __forceinline__ float bf2f(unsigned short h) {
  return __uint_as_float(((unsigned)h) << 16);
}
__device__ __forceinline__ float sigmoidf_(float x) {
  return 1.0f / (1.0f + __expf(-x));
}

// CDNA5 async global->LDS copy (ASYNCcnt-tracked), 16 bytes per lane.
__device__ __forceinline__ void async_copy16(unsigned lds_off, const void* gptr) {
  asm volatile("global_load_async_to_lds_b128 %0, %1, off"
               :: "v"(lds_off), "v"(gptr) : "memory");
}
__device__ __forceinline__ void wait_async0() {
  asm volatile("s_wait_asynccnt 0" ::: "memory");
}

// ---------------------------------------------------------------------------
// adjT[m*N+n] = adj[n*N+m]
__global__ void transpose_kernel(const float* __restrict__ a, float* __restrict__ at) {
  __shared__ float t[16][17];
  int x = blockIdx.x * 16 + threadIdx.x;
  int y = blockIdx.y * 16 + threadIdx.y;
  t[threadIdx.y][threadIdx.x] = a[y * N_ + x];
  __syncthreads();
  int xo = blockIdx.y * 16 + threadIdx.x;
  int yo = blockIdx.x * 16 + threadIdx.y;
  at[yo * N_ + xo] = t[threadIdx.x][threadIdx.y];
}

// ---------------------------------------------------------------------------
// AT[m*N+n] = softmax_m(relu(E1[n] . E2[m]))  (transposed dynamic adjacency)
__global__ __launch_bounds__(256)
void dynadj_kernel(const float* __restrict__ E1, const float* __restrict__ E2,
                   float* __restrict__ AT) {
  __shared__ float e1[D_];
  __shared__ float red[256];
  int n = blockIdx.x;
  int t = threadIdx.x;
  if (t < D_) e1[t] = E1[n * D_ + t];
  __syncthreads();
  float vals[2];
  float vmax = -1e30f;
  #pragma unroll
  for (int j = 0; j < 2; ++j) {
    int m = t + j * 256;
    float s = 0.0f;
    #pragma unroll
    for (int d = 0; d < D_; ++d) s += e1[d] * E2[m * D_ + d];
    s = fmaxf(s, 0.0f);
    vals[j] = s;
    vmax = fmaxf(vmax, s);
  }
  red[t] = vmax; __syncthreads();
  for (int s2 = 128; s2 > 0; s2 >>= 1) { if (t < s2) red[t] = fmaxf(red[t], red[t + s2]); __syncthreads(); }
  vmax = red[0]; __syncthreads();
  float lsum = 0.0f;
  #pragma unroll
  for (int j = 0; j < 2; ++j) { vals[j] = __expf(vals[j] - vmax); lsum += vals[j]; }
  red[t] = lsum; __syncthreads();
  for (int s2 = 128; s2 > 0; s2 >>= 1) { if (t < s2) red[t] += red[t + s2]; __syncthreads(); }
  float inv = 1.0f / red[0];
  #pragma unroll
  for (int j = 0; j < 2; ++j) AT[(size_t)(t + j * 256) * N_ + n] = vals[j] * inv;
}

// ---------------------------------------------------------------------------
// Augmented gate matrices, 2 variants (double-buffered msg slot), bf16:
// Uaug[v][g][o][k]: k<128: U[g][o][k];
//   k in [128+16v, 128+16v+9): W[g][o][k-(128+16v)]; k == 128+16v+9: bias[g][o];
//   else 0  (the other 16-row slot has zero coefficients -> safe async prefetch)
__global__ void build_uaug_kernel(const float* __restrict__ U, const float* __restrict__ W,
                                  const float* __restrict__ bias,
                                  unsigned short* __restrict__ Uaug) {
  int i = blockIdx.x * 256 + threadIdx.x;       // 2*3*128*KAUG_
  if (i >= 2 * 3 * E_ * KAUG_) return;
  int k = i % KAUG_;
  int go = (i / KAUG_) % (3 * E_);               // g*128 + o
  int v = i / (3 * E_ * KAUG_);                  // msg-slot variant
  int base = 128 + v * 16;
  float val = 0.0f;
  if (k < 128)               val = U[go * 128 + k];
  else if (k >= base && k < base + 9)  val = W[go * 9 + (k - base)];
  else if (k == base + 9)    val = bias[go];
  Uaug[i] = f2bf(val);
}

// ---------------------------------------------------------------------------
// Message precompute.
//  mode<0 (forward, grid.y = H*B): write packed bf16 msgT[hb][m][16]:
//    ch 0..2 = x, 3..5 = x@adj, 6..8 = x@A_dyn, 9 = 1.0 (bias row), 10..15 = 0
//  mode>=0 (backward, grid.y = B, h = mode): write f32 msgF[b][9][N]
__global__ __launch_bounds__(128)
void msg_kernel(const float* __restrict__ x, const float* __restrict__ adjT,
                const float* __restrict__ AT, unsigned short* __restrict__ msgT,
                float* __restrict__ msgF, int h_override) {
  __shared__ float xs[N_ * C_];
  int hb = blockIdx.y;
  int h, b;
  if (h_override >= 0) { h = h_override; b = hb; }
  else { h = hb >> 4; b = hb & 15; }
  int t = threadIdx.x;
  const float* xp = x + ((size_t)(b * H_ + h)) * N_ * C_;
  for (int i = t; i < N_ * C_; i += 128) xs[i] = xp[i];
  __syncthreads();
  int m = blockIdx.x * 128 + t;
  float ap0 = 0, ap1 = 0, ap2 = 0, ad0 = 0, ad1 = 0, ad2 = 0;
  const float* arow = adjT + (size_t)m * N_;
  const float* frow = AT + (size_t)m * N_;
  for (int n = 0; n < N_; ++n) {
    float a = arow[n], f = frow[n];
    float x0 = xs[n * 3 + 0], x1 = xs[n * 3 + 1], x2 = xs[n * 3 + 2];
    ap0 += a * x0; ap1 += a * x1; ap2 += a * x2;
    ad0 += f * x0; ad1 += f * x1; ad2 += f * x2;
  }
  if (h_override < 0) {
    unsigned short vals[16];
    vals[0] = f2bf(xs[m * 3 + 0]); vals[1] = f2bf(xs[m * 3 + 1]); vals[2] = f2bf(xs[m * 3 + 2]);
    vals[3] = f2bf(ap0); vals[4] = f2bf(ap1); vals[5] = f2bf(ap2);
    vals[6] = f2bf(ad0); vals[7] = f2bf(ad1); vals[8] = f2bf(ad2);
    vals[9] = f2bf(1.0f);
    #pragma unroll
    for (int c = 10; c < 16; ++c) vals[c] = 0;
    v4u p0, p1;
    #pragma unroll
    for (int j = 0; j < 4; ++j) {
      p0[j] = (unsigned)vals[2 * j] | ((unsigned)vals[2 * j + 1] << 16);
      p1[j] = (unsigned)vals[8 + 2 * j] | ((unsigned)vals[9 + 2 * j] << 16);
    }
    v4u* dst = reinterpret_cast<v4u*>(msgT + ((size_t)hb * N_ + m) * 16);
    dst[0] = p0; dst[1] = p1;
  } else {
    float* mp = msgF + (size_t)hb * 9 * N_;
    mp[0 * N_ + m] = xs[m * 3 + 0];
    mp[1 * N_ + m] = xs[m * 3 + 1];
    mp[2 * N_ + m] = xs[m * 3 + 2];
    mp[3 * N_ + m] = ap0; mp[4 * N_ + m] = ap1; mp[5 * N_ + m] = ap2;
    mp[6 * N_ + m] = ad0; mp[7 * N_ + m] = ad1; mp[8 * N_ + m] = ad2;
  }
}

// ---------------------------------------------------------------------------
// WMMA fragment helpers (CDNA5 bf16 16x16x32, wave32)
__device__ __forceinline__ v16bf load_a_frag(const unsigned short* __restrict__ Ug,
                                             int row, int ks, int lane) {
  int kb = ks * 32 + ((lane >> 4) & 1) * 8;
  const unsigned short* p = Ug + (size_t)row * KAUG_ + kb;
  FragBF16 f;
  f.q[0] = *reinterpret_cast<const v4u*>(p);
  f.q[1] = *reinterpret_cast<const v4u*>(p + 16);
  return f.v;
}
__device__ __forceinline__ v16bf load_b_frag(const unsigned short* __restrict__ tile,
                                             int nt, int ks, int lane) {
  int n = nt * 16 + (lane & 15);
  int kb = ks * 32 + ((lane >> 4) & 1) * 8;
  const unsigned short* p = tile + n * KAUG_ + kb;
  FragBF16 f;
  f.q[0] = *reinterpret_cast<const v4u*>(p);
  f.q[1] = *reinterpret_cast<const v4u*>(p + 16);
  return f.v;
}

// Async-load one 16-row msg slot (region reg in {0,1}) for this block's columns.
__device__ __forceinline__ void stage_msg(unsigned short* tile,
                                          const unsigned short* __restrict__ msgTh,
                                          int b, int ncol0, int t, int reg) {
  if (t < 128) {
    int n = t >> 1, part = t & 1;
    async_copy16((unsigned)(size_t)&tile[n * KAUG_ + 128 + reg * 16 + part * 8],
                 msgTh + ((size_t)b * N_ + ncol0 + n) * 16 + part * 8);
  }
}

// ---------------------------------------------------------------------------
// Persistent full-scan kernel: the GRU recurrence is column-local, so each
// workgroup owns 64 columns of one batch and runs all H=64 steps with the
// state tile resident in LDS. Per step only the 16 msg rows are streamed in
// (async, double-buffered into the inactive slot while WMMAs run).
//   step h (slot s=h&1, Uaug variant s):
//     acc0/acc1 = Uaug0/Uaug1 @ [c ; msgslots]  (r,u gates)
//     r,u = sigmoid; state rows <- r*c (in LDS)
//     acc2 = Uaug2 @ [r*c ; msgslots];  c' = u*c + (1-u)*tanh(acc2) -> state rows
__global__ __launch_bounds__(256)
void ibn_scan_kernel(const unsigned short* __restrict__ Uaug,   // [2][3][E][KAUG] bf16
                     const unsigned short* __restrict__ msgT,   // [H][B][N][16] bf16
                     unsigned short* __restrict__ hOut) {       // [B][N][E] bf16
  __shared__ __align__(16) unsigned short tile[64 * KAUG_];
  int t = threadIdx.x;
  int b = blockIdx.y;
  int ncol0 = blockIdx.x * 64;

  // --- prologue: zero state rows 0..63 only + slot1; async-load msg(h=0) into slot0 ---
  v4u z4 = {0, 0, 0, 0};
  #pragma unroll
  for (int idx = t; idx < 512; idx += 256) {
    int n = idx >> 3, c8 = idx & 7;
    *reinterpret_cast<v4u*>(&tile[n * KAUG_ + c8 * 8]) = z4;
  }
  if (t < 128) {
    int n = t >> 1, part = t & 1;
    *reinterpret_cast<v4u*>(&tile[n * KAUG_ + 144 + part * 8]) = z4;
  }
  stage_msg(tile, msgT, b, ncol0, t, 0);
  wait_async0();
  __syncthreads();

  int w = t >> 5, lane = t & 31;
  int lhalf = (lane >> 4) & 1;
  int lmod = lane & 15;
  int orow = w * 16 + lhalf * 8;     // this lane's 8 consecutive output rows
  v8f z8 = {0, 0, 0, 0, 0, 0, 0, 0};
  v4u outpk[4];

  for (int h = 0; h < H_; ++h) {
    const unsigned short* Ua = Uaug + (size_t)(h & 1) * 3 * E_ * KAUG_;

    // prefetch next step's msg into the inactive slot (zero A-coeffs there)
    if (h + 1 < H_)
      stage_msg(tile, msgT + (size_t)(h + 1) * B_ * N_ * 16, b, ncol0, t, (h + 1) & 1);

    // --- GEMM gates r (0) and u (1) ---
    v8f acc0[4], acc1[4];
    #pragma unroll
    for (int nt = 0; nt < 4; ++nt) { acc0[nt] = z8; acc1[nt] = z8; }
    #pragma unroll
    for (int ks = 0; ks < 5; ++ks) {
      v16bf a0 = load_a_frag(Ua + 0 * E_ * KAUG_, w * 16 + lmod, ks, lane);
      v16bf a1 = load_a_frag(Ua + 1 * E_ * KAUG_, w * 16 + lmod, ks, lane);
      #pragma unroll
      for (int nt = 0; nt < 4; ++nt) {
        v16bf bq = load_b_frag(tile, nt, ks, lane);
        acc0[nt] = __builtin_amdgcn_wmma_f32_16x16x32_bf16(false, a0, false, bq,
                                                           (short)0, acc0[nt], false, false);
        acc1[nt] = __builtin_amdgcn_wmma_f32_16x16x32_bf16(false, a1, false, bq,
                                                           (short)0, acc1[nt], false, false);
      }
    }
    __syncthreads();   // all waves done reading state rows

    // --- r,u; overwrite state rows with r*c (per-lane 16B, disjoint slices) ---
    float ureg[4][8], cvreg[4][8];
    #pragma unroll
    for (int nt = 0; nt < 4; ++nt) {
      int nl = nt * 16 + lmod;
      unsigned short* tp = &tile[nl * KAUG_ + orow];
      v4u cpk = *reinterpret_cast<const v4u*>(tp);
      unsigned short rcb[8];
      #pragma unroll
      for (int i = 0; i < 8; ++i) {
        float cv = bf2f((unsigned short)((cpk[i >> 1] >> ((i & 1) * 16)) & 0xffffu));
        float r = sigmoidf_(acc0[nt][i]);
        float u = sigmoidf_(acc1[nt][i]);
        ureg[nt][i] = u;
        cvreg[nt][i] = cv;
        rcb[i] = f2bf(r * cv);
      }
      v4u pk;
      #pragma unroll
      for (int j = 0; j < 4; ++j)
        pk[j] = (unsigned)rcb[2 * j] | ((unsigned)rcb[2 * j + 1] << 16);
      *reinterpret_cast<v4u*>(tp) = pk;
    }
    __syncthreads();   // r*c tile complete

    // --- GEMM gate g (2) on [r*c ; msg ; 1] ---
    v8f acc2[4];
    #pragma unroll
    for (int nt = 0; nt < 4; ++nt) acc2[nt] = z8;
    #pragma unroll
    for (int ks = 0; ks < 5; ++ks) {
      v16bf a2 = load_a_frag(Ua + 2 * E_ * KAUG_, w * 16 + lmod, ks, lane);
      #pragma unroll
      for (int nt = 0; nt < 4; ++nt) {
        v16bf bq = load_b_frag(tile, nt, ks, lane);
        acc2[nt] = __builtin_amdgcn_wmma_f32_16x16x32_bf16(false, a2, false, bq,
                                                           (short)0, acc2[nt], false, false);
      }
    }
    __syncthreads();   // all waves done reading r*c rows

    // --- state update: c' = u*c + (1-u)*tanh(acc2) back into state rows ---
    #pragma unroll
    for (int nt = 0; nt < 4; ++nt) {
      int nl = nt * 16 + lmod;
      unsigned short cb[8];
      #pragma unroll
      for (int i = 0; i < 8; ++i) {
        float g = tanhf(acc2[nt][i]);
        float u = ureg[nt][i];
        float cv = cvreg[nt][i];
        cb[i] = f2bf(u * cv + (1.0f - u) * g);
      }
      v4u pk;
      #pragma unroll
      for (int j = 0; j < 4; ++j)
        pk[j] = (unsigned)cb[2 * j] | ((unsigned)cb[2 * j + 1] << 16);
      *reinterpret_cast<v4u*>(&tile[nl * KAUG_ + orow]) = pk;
      outpk[nt] = pk;
    }
    wait_async0();     // prefetched msg slot resident
    __syncthreads();   // state rows consistent for next step
  }

  // --- epilogue: final hidden state to global (b128 stores) ---
  #pragma unroll
  for (int nt = 0; nt < 4; ++nt) {
    int col = ncol0 + nt * 16 + lmod;
    *reinterpret_cast<v4u*>(hOut + ((size_t)b * N_ + col) * E_ + orow) = outpk[nt];
  }
}

// ---------------------------------------------------------------------------
// Backward cell first step with c=0: h_back = (1-sig(gx1+b1)) * tanh(gx2+b2)
// Writes transposed layout hbT[b][n][o].
__global__ __launch_bounds__(256)
void hback_kernel(const float* __restrict__ msgb, const float* __restrict__ Wb,
                  const float* __restrict__ bb, float* __restrict__ hbT) {
  int idx = blockIdx.x * 256 + threadIdx.x;   // B*E*N
  int n = idx & (N_ - 1);
  int o = (idx >> 9) & (E_ - 1);
  int b = idx >> 16;
  const float* mp = msgb + (size_t)b * 9 * N_;
  float g1 = 0, g2 = 0;
  #pragma unroll
  for (int c = 0; c < 9; ++c) {
    float m = mp[(size_t)c * N_ + n];
    g1 += Wb[(1 * E_ + o) * 9 + c] * m;
    g2 += Wb[(2 * E_ + o) * 9 + c] * m;
  }
  float u = sigmoidf_(g1 + bb[E_ + o]);
  float g = tanhf(g2 + bb[2 * E_ + o]);
  hbT[((size_t)b * N_ + n) * E_ + o] = (1.0f - u) * g;
}

// ---------------------------------------------------------------------------
// Decoder: dw2[o,e] = sum_l dec_w[o,l,e]; h[o] = dec_b[o] + sum_e lt[e]*dw2[o,e];
// out[j] = out_b[j] + sum_o out_w[j,o]*h[o].  hfT bf16 / hbT f32, both [B][N][E].
__global__ __launch_bounds__(64)
void decoder_kernel(const unsigned short* __restrict__ hfT, const float* __restrict__ hbT,
                    const float* __restrict__ dec_w, const float* __restrict__ dec_b,
                    const float* __restrict__ out_w, const float* __restrict__ out_b,
                    float* __restrict__ out) {
  __shared__ float dcomb[LOUT_ * 2 * E_];
  int idx = blockIdx.x * 64 + threadIdx.x;    // B*N
  for (int i = threadIdx.x; i < LOUT_ * 2 * E_; i += 64) {
    int o = i >> 8, e = i & 255;
    dcomb[i] = dec_w[(o * 2 + 0) * 256 + e] + dec_w[(o * 2 + 1) * 256 + e];
  }
  __syncthreads();
  int n = idx & (N_ - 1);
  int b = idx >> 9;
  const unsigned short* hf = hfT + ((size_t)b * N_ + n) * E_;
  const float* hb = hbT + ((size_t)b * N_ + n) * E_;
  float ho[LOUT_];
  #pragma unroll
  for (int o = 0; o < LOUT_; ++o) ho[o] = dec_b[o];
  for (int e = 0; e < E_; ++e) {
    float vf = bf2f(hf[e]);
    float vb = hb[e];
    #pragma unroll
    for (int o = 0; o < LOUT_; ++o)
      ho[o] += vf * dcomb[o * 256 + e] + vb * dcomb[o * 256 + E_ + e];
  }
  #pragma unroll
  for (int j = 0; j < LOUT_; ++j) {
    float s = out_b[j];
    #pragma unroll
    for (int o = 0; o < LOUT_; ++o) s += out_w[j * LOUT_ + o] * ho[o];
    out[((size_t)b * LOUT_ + j) * N_ + n] = s;
  }
}

// ---------------------------------------------------------------------------
extern "C" void kernel_launch(void* const* d_in, const int* in_sizes, int n_in,
                              void* d_out, int out_size, void* d_ws, size_t ws_size,
                              hipStream_t stream) {
  const float* x     = (const float*)d_in[0];
  const float* adj   = (const float*)d_in[1];
  const float* Wf    = (const float*)d_in[2];
  const float* Uf    = (const float*)d_in[3];
  const float* bfv   = (const float*)d_in[4];
  const float* E1f   = (const float*)d_in[5];
  const float* E2f   = (const float*)d_in[6];
  const float* Wb    = (const float*)d_in[7];
  const float* bb    = (const float*)d_in[9];
  const float* E1b   = (const float*)d_in[10];
  const float* E2b   = (const float*)d_in[11];
  const float* dec_w = (const float*)d_in[12];
  const float* dec_b = (const float*)d_in[13];
  const float* out_w = (const float*)d_in[14];
  const float* out_b = (const float*)d_in[15];
  float* out = (float*)d_out;

  char* ws = (char*)d_ws;
  size_t off = 0;
  auto alloc = [&](size_t bytes) -> char* {
    char* p = ws + off;
    off = (off + bytes + 255) & ~(size_t)255;
    return p;
  };
  float* adjT = (float*)alloc((size_t)N_ * N_ * 4);
  float* AfT  = (float*)alloc((size_t)N_ * N_ * 4);
  float* AbT  = (float*)alloc((size_t)N_ * N_ * 4);
  unsigned short* Uaug = (unsigned short*)alloc((size_t)2 * 3 * E_ * KAUG_ * 2);
  unsigned short* msgT = (unsigned short*)alloc((size_t)H_ * B_ * N_ * 16 * 2);
  float* msgB = (float*)alloc((size_t)B_ * 9 * N_ * 4);
  unsigned short* hfwdT = (unsigned short*)alloc((size_t)B_ * N_ * E_ * 2);
  float* hbT  = (float*)alloc((size_t)B_ * N_ * E_ * 4);

  // --- precompute ---
  transpose_kernel<<<dim3(N_ / 16, N_ / 16), dim3(16, 16), 0, stream>>>(adj, adjT);
  dynadj_kernel<<<N_, 256, 0, stream>>>(E1f, E2f, AfT);
  dynadj_kernel<<<N_, 256, 0, stream>>>(E1b, E2b, AbT);
  build_uaug_kernel<<<(2 * 3 * E_ * KAUG_ + 255) / 256, 256, 0, stream>>>(Uf, Wf, bfv, Uaug);
  msg_kernel<<<dim3(N_ / 128, H_ * B_), 128, 0, stream>>>(x, adjT, AfT, msgT, nullptr, -1);
  msg_kernel<<<dim3(N_ / 128, B_), 128, 0, stream>>>(x, adjT, AbT, nullptr, msgB, H_ - 1);

  // --- full recurrence in ONE persistent kernel (state LDS-resident) ---
  ibn_scan_kernel<<<dim3(N_ / 64, B_), 256, 0, stream>>>(Uaug, msgT, hfwdT);

  // --- backward first step + decoder ---
  hback_kernel<<<(B_ * E_ * N_) / 256, 256, 0, stream>>>(msgB, Wb, bb, hbT);
  decoder_kernel<<<(B_ * N_) / 64, 64, 0, stream>>>(hfwdT, hbT, dec_w, dec_b,
                                                    out_w, out_b, out);
}
